// MHA_33998961115807
// MI455X (gfx1250) — matrix operand order repaired and checked by reading, not compiled, not run on hardware
//
#include <hip/hip_runtime.h>
#include <cstdint>

typedef __attribute__((ext_vector_type(16))) __bf16 bf16x16;
typedef __attribute__((ext_vector_type(8)))  float  f32x8;

union FragB { uint32_t u[8]; bf16x16 v; };
union FragC { f32x8 v; float f[8]; };

constexpr int kB   = 2;
constexpr int kN   = 2048;
constexpr int kC   = 1024;
constexpr int kNH  = 16;
constexpr int kHD  = 64;
constexpr int kM   = kB * kN;     // 4096 GEMM rows
constexpr int kQKVN = 3 * kC;     // 3072

// ---- helpers -------------------------------------------------------------

__device__ __forceinline__ unsigned short f2bf(float f) {
  uint32_t u = __float_as_uint(f);
  u += 0x7fffu + ((u >> 16) & 1u);       // round-to-nearest-even
  return (unsigned short)(u >> 16);
}

__device__ __forceinline__ uint32_t ldpair(const unsigned short* p) {
  return *reinterpret_cast<const uint32_t*>(p);   // K0 always even -> 4B aligned
}

// A-fragment 16x32 bf16: lane half selects K+8; VGPR j -> K pairs
__device__ __forceinline__ int aK(int j, int half) {
  return ((j >> 2) << 4) + (half << 3) + ((j & 3) << 1);
}
// B-fragment 32x16 bf16: lane half selects K+8; VGPR j -> K pairs (same as A)
__device__ __forceinline__ int bK(int j, int half) {
  return ((j >> 2) << 4) + (half << 3) + ((j & 3) << 1);
}

__device__ __forceinline__ void loadAfrag(FragB& f, const unsigned short* row,
                                          int kc, int half) {
#pragma unroll
  for (int j = 0; j < 8; ++j) f.u[j] = ldpair(row + kc + aK(j, half));
}

__device__ __forceinline__ FragC wmma_bf16(const FragB& a, const FragB& b, FragC c) {
  c.v = __builtin_amdgcn_wmma_f32_16x16x32_bf16(false, a.v, false, b.v,
                                                (short)0, c.v, false, false);
  return c;
}

// ---- stage 0: f32 -> bf16 ------------------------------------------------

__global__ void __launch_bounds__(256)
cvt_bf16_kernel(const float* __restrict__ in, unsigned short* __restrict__ out, int n) {
  int i = blockIdx.x * 256 + threadIdx.x;
  if (i < n) out[i] = f2bf(in[i]);
}

// ---- shared 32(M) x 64(N) tile GEMM core, double-buffered ---------------
// A [M x K] row-major bf16, B_wmma(k,n) = W[n,k] with W [Nd x K] row-major.

__device__ __forceinline__ void gemm_tile_32x64(const unsigned short* __restrict__ Am,
                                                const unsigned short* __restrict__ Wm,
                                                int K, int row0, int col0,
                                                int lane, FragC acc[2][4]) {
  const int half = lane >> 4, l16 = lane & 15;
  const unsigned short* ar0 = Am + (size_t)(row0 + l16) * K;
  const unsigned short* ar1 = Am + (size_t)(row0 + 16 + l16) * K;
  const unsigned short* br[4];
#pragma unroll
  for (int t = 0; t < 4; ++t) br[t] = Wm + (size_t)(col0 + t * 16 + l16) * K;

#pragma unroll
  for (int m = 0; m < 2; ++m)
#pragma unroll
    for (int t = 0; t < 4; ++t)
#pragma unroll
      for (int r = 0; r < 8; ++r) acc[m][t].f[r] = 0.f;

  FragB a[2][2], b[2][4];
  loadAfrag(a[0][0], ar0, 0, half);
  loadAfrag(a[0][1], ar1, 0, half);
#pragma unroll
  for (int t = 0; t < 4; ++t)
#pragma unroll
    for (int j = 0; j < 8; ++j) b[0][t].u[j] = ldpair(br[t] + bK(j, half));

  int cur = 0;
  for (int kc = 0; kc < K; kc += 32) {
    const int nxt = cur ^ 1;
    if (kc + 32 < K) {                       // prefetch next chunk's fragments
      loadAfrag(a[nxt][0], ar0, kc + 32, half);
      loadAfrag(a[nxt][1], ar1, kc + 32, half);
#pragma unroll
      for (int t = 0; t < 4; ++t)
#pragma unroll
        for (int j = 0; j < 8; ++j)
          b[nxt][t].u[j] = ldpair(br[t] + kc + 32 + bK(j, half));
      if (kc + 64 < K) {                     // L2 prefetch one more chunk out
        __builtin_prefetch(ar0 + kc + 64, 0, 3);
        __builtin_prefetch(ar1 + kc + 64, 0, 3);
#pragma unroll
        for (int t = 0; t < 4; ++t) __builtin_prefetch(br[t] + kc + 64, 0, 3);
      }
    }
#pragma unroll
    for (int t = 0; t < 4; ++t) {
      acc[0][t] = wmma_bf16(a[cur][0], b[cur][t], acc[0][t]);
      acc[1][t] = wmma_bf16(a[cur][1], b[cur][t], acc[1][t]);
    }
    cur = nxt;
  }
}

// ---- stage 1: QKV = x @ Wqkv^T + b, scattered per head -------------------

__global__ void __launch_bounds__(256)
qkv_gemm_kernel(const unsigned short* __restrict__ xb,
                const unsigned short* __restrict__ wb,
                const float* __restrict__ bias,
                unsigned short* __restrict__ qb,
                unsigned short* __restrict__ kbuf,
                unsigned short* __restrict__ vtb) {
  const int lane = threadIdx.x & 31;
  const int wave = (blockIdx.x * blockDim.x + threadIdx.x) >> 5;
  const int nTiles = kQKVN / 64;                 // 48
  const int mt = wave / nTiles;
  const int nt = wave % nTiles;
  if (mt >= kM / 32) return;
  const int row0 = mt * 32, col0 = nt * 64;
  const int half = lane >> 4, l16 = lane & 15;

  FragC acc[2][4];
  gemm_tile_32x64(xb, wb, kC, row0, col0, lane, acc);

#pragma unroll
  for (int m = 0; m < 2; ++m) {
#pragma unroll
    for (int t = 0; t < 4; ++t) {
#pragma unroll
      for (int r = 0; r < 8; ++r) {
        int grow = row0 + m * 16 + r + (half << 3);
        int gcol = col0 + t * 16 + l16;
        float v = acc[m][t].f[r] + bias[gcol];
        int b_  = grow >> 11;                    // / kN
        int rw  = grow & (kN - 1);
        int sec = gcol / kC;
        int c   = gcol - sec * kC;
        int h   = c >> 6;
        int d   = c & 63;
        int bh  = b_ * kNH + h;
        unsigned short bv = f2bf(v);
        if (sec == 0)      qb  [((size_t)bh * kN + rw) * kHD + d] = bv;
        else if (sec == 1) kbuf[((size_t)bh * kN + rw) * kHD + d] = bv;
        else               vtb [((size_t)bh * kHD + d) * kN + rw] = bv;
      }
    }
  }
}

// ---- stage 2: causal flash attention, one wave (=workgroup) per Q tile ---

__global__ void __launch_bounds__(32)
attn_kernel(const unsigned short* __restrict__ qb,
            const unsigned short* __restrict__ kb,
            const unsigned short* __restrict__ vtb,
            unsigned short* __restrict__ yb) {
  __shared__ unsigned short Pl[16 * 32];     // P tile, row-major bf16
  const int lane = threadIdx.x & 31;
  const int half = lane >> 4, l16 = lane & 15;
  const int wid  = blockIdx.x;
  const int bh   = wid >> 7;                 // 128 q-tiles per (b,h)
  const int mt   = wid & 127;
  const int row0 = mt * 16;
  const int b_   = bh >> 4;
  const int h    = bh & 15;
  const float scale = 0.125f;                // hd^-0.5

  const unsigned short* qbase = qb  + (size_t)bh * kN * kHD;
  const unsigned short* kbase = kb  + (size_t)bh * kN * kHD;
  const unsigned short* vbase = vtb + (size_t)bh * kHD * kN;

  FragB qf0, qf1;                            // Q 16x64 = two k=32 A-frags
#pragma unroll
  for (int j = 0; j < 8; ++j) {
    int k0 = aK(j, half);
    const unsigned short* qr = qbase + (size_t)(row0 + l16) * kHD;
    qf0.u[j] = ldpair(qr + k0);
    qf1.u[j] = ldpair(qr + 32 + k0);
  }

  float om[8], osum[8];
  FragC o[4];
#pragma unroll
  for (int r = 0; r < 8; ++r) { om[r] = -1e30f; osum[r] = 0.f; }
#pragma unroll
  for (int t = 0; t < 4; ++t)
#pragma unroll
    for (int r = 0; r < 8; ++r) o[t].f[r] = 0.f;

  const int nchunks = (row0 + 15) / 32 + 1;  // causal early exit
  for (int kc = 0; kc < nchunks; ++kc) {
    const int key0 = kc * 32;

    // K fragments for both 16-key tiles (d 0..31 / 32..63 per tile)
    FragB bk[2][2];
#pragma unroll
    for (int t = 0; t < 2; ++t) {
      int nkey = key0 + t * 16 + l16;
      const unsigned short* kr = kbase + (size_t)nkey * kHD;
#pragma unroll
      for (int j = 0; j < 8; ++j) {
        int k0 = bK(j, half);
        bk[t][0].u[j] = ldpair(kr + k0);
        bk[t][1].u[j] = ldpair(kr + 32 + k0);
      }
    }
    // V fragments depend only on key0 — issue now, consumed after softmax
    FragB vf[4];
#pragma unroll
    for (int t4 = 0; t4 < 4; ++t4) {
      int d = t4 * 16 + l16;
#pragma unroll
      for (int j = 0; j < 8; ++j)
        vf[t4].u[j] = ldpair(vbase + (size_t)d * kN + key0 + bK(j, half));
    }

    float st[2][8];
#pragma unroll
    for (int t = 0; t < 2; ++t) {
      FragC s;
#pragma unroll
      for (int r = 0; r < 8; ++r) s.f[r] = 0.f;
      s = wmma_bf16(qf0, bk[t][0], s);
      s = wmma_bf16(qf1, bk[t][1], s);
#pragma unroll
      for (int r = 0; r < 8; ++r) {
        int grow = row0 + r + (half << 3);
        int gcol = key0 + t * 16 + l16;
        st[t][r] = (gcol > grow) ? -1e30f : s.f[r] * scale;   // causal mask
      }
    }

    __syncthreads();   // WAR: previous iteration's P loads vs new stores
#pragma unroll
    for (int r = 0; r < 8; ++r) {
      float cm = fmaxf(st[0][r], st[1][r]);
#pragma unroll
      for (int off = 1; off < 16; off <<= 1) cm = fmaxf(cm, __shfl_xor(cm, off, 32));
      float nm   = fmaxf(om[r], cm);
      float corr = __expf(om[r] - nm);
      om[r] = nm;
      float p0 = __expf(st[0][r] - nm);
      float p1 = __expf(st[1][r] - nm);
      float rs = p0 + p1;
#pragma unroll
      for (int off = 1; off < 16; off <<= 1) rs += __shfl_xor(rs, off, 32);
      osum[r] = osum[r] * corr + rs;
#pragma unroll
      for (int t4 = 0; t4 < 4; ++t4) o[t4].f[r] *= corr;
      int prow = r + (half << 3);
      Pl[prow * 32 + l16]      = f2bf(p0);
      Pl[prow * 32 + 16 + l16] = f2bf(p1);
    }
    __syncthreads();   // RAW: P stores before A-fragment reload

    FragB pa;
#pragma unroll
    for (int j = 0; j < 8; ++j)
      pa.u[j] = *reinterpret_cast<const uint32_t*>(&Pl[l16 * 32 + aK(j, half)]);

#pragma unroll
    for (int t4 = 0; t4 < 4; ++t4)
      o[t4] = wmma_bf16(pa, vf[t4], o[t4]);
  }

  // normalize + store attn-out directly in [B,N,C] layout (c = h*64+d)
#pragma unroll
  for (int t4 = 0; t4 < 4; ++t4) {
#pragma unroll
    for (int r = 0; r < 8; ++r) {
      int grow = row0 + r + (half << 3);
      int d = t4 * 16 + l16;
      float val = o[t4].f[r] / osum[r];
      yb[((size_t)(b_ * kN + grow)) * kC + h * kHD + d] = f2bf(val);
    }
  }
}

// ---- stage 3: out = y @ Wproj^T + b (fp32 output) ------------------------

__global__ void __launch_bounds__(256)
proj_gemm_kernel(const unsigned short* __restrict__ yb,
                 const unsigned short* __restrict__ wb,
                 const float* __restrict__ bias,
                 float* __restrict__ out) {
  const int lane = threadIdx.x & 31;
  const int wave = (blockIdx.x * blockDim.x + threadIdx.x) >> 5;
  const int nTiles = kC / 64;                    // 16
  const int mt = wave / nTiles;
  const int nt = wave % nTiles;
  if (mt >= kM / 32) return;
  const int row0 = mt * 32, col0 = nt * 64;
  const int half = lane >> 4, l16 = lane & 15;

  FragC acc[2][4];
  gemm_tile_32x64(yb, wb, kC, row0, col0, lane, acc);

#pragma unroll
  for (int m = 0; m < 2; ++m) {
#pragma unroll
    for (int t = 0; t < 4; ++t) {
#pragma unroll
      for (int r = 0; r < 8; ++r) {
        int grow = row0 + m * 16 + r + (half << 3);
        int gcol = col0 + t * 16 + l16;
        out[(size_t)grow * kC + gcol] = acc[m][t].f[r] + bias[gcol];
      }
    }
  }
}

// ---- host glue -----------------------------------------------------------

extern "C" void kernel_launch(void* const* d_in, const int* in_sizes, int n_in,
                              void* d_out, int out_size, void* d_ws, size_t ws_size,
                              hipStream_t stream) {
  const float* x     = (const float*)d_in[0];
  // d_in[1] = causal mask (bool) — triu(k=1), applied analytically in attn_kernel
  const float* Wqkv  = (const float*)d_in[2];
  const float* bqkv  = (const float*)d_in[3];
  const float* Wproj = (const float*)d_in[4];
  const float* bproj = (const float*)d_in[5];
  float* out = (float*)d_out;

  char* ws = (char*)d_ws;
  const size_t nX  = (size_t)kM * kC;              // 4 Mi elems
  const size_t nWq = (size_t)kQKVN * kC;           // 3 Mi
  const size_t nWp = (size_t)kC * kC;              // 1 Mi
  const size_t nHd = (size_t)kB * kNH * kN * kHD;  // 4 Mi (per Q/K/V)

  unsigned short* xb  = (unsigned short*)(ws);
  unsigned short* wqb = (unsigned short*)(ws + 2 * nX);
  unsigned short* wpb = (unsigned short*)(ws + 2 * (nX + nWq));
  unsigned short* qb  = (unsigned short*)(ws + 2 * (nX + nWq + nWp));
  unsigned short* kb  = (unsigned short*)(ws + 2 * (nX + nWq + nWp + nHd));
  unsigned short* vtb = (unsigned short*)(ws + 2 * (nX + nWq + nWp + 2 * nHd));
  unsigned short* yb  = (unsigned short*)(ws + 2 * (nX + nWq + nWp + 3 * nHd));

  cvt_bf16_kernel<<<(int)((nX  + 255) / 256), 256, 0, stream>>>(x,     xb,  (int)nX);
  cvt_bf16_kernel<<<(int)((nWq + 255) / 256), 256, 0, stream>>>(Wqkv,  wqb, (int)nWq);
  cvt_bf16_kernel<<<(int)((nWp + 255) / 256), 256, 0, stream>>>(Wproj, wpb, (int)nWp);

  // QKV GEMM: (4096/32) * (3072/64) = 6144 waves, 8 waves/block
  qkv_gemm_kernel<<<6144 * 32 / 256, 256, 0, stream>>>(xb, wqb, bqkv, qb, kb, vtb);

  // attention: one wave per (b,h,16-row tile) = 2*16*128 = 4096 workgroups
  attn_kernel<<<kB * kNH * (kN / 16), 32, 0, stream>>>(qb, kb, vtb, yb);

  // projection: (4096/32) * (1024/64) = 2048 waves
  proj_gemm_kernel<<<2048 * 32 / 256, 256, 0, stream>>>(yb, wpb, bproj, out);
}
